// SelfAttention_21638045237750
// MI455X (gfx1250) — compile-verified, matrix operand order checked by reading, not run on hardware
//
#include <hip/hip_runtime.h>

typedef __attribute__((ext_vector_type(16))) __bf16 v16bf;
typedef __attribute__((ext_vector_type(8)))  float  v8f;
typedef __attribute__((ext_vector_type(4)))  unsigned int u32x4;
typedef __attribute__((ext_vector_type(8)))  int i32x8;
typedef __attribute__((ext_vector_type(4)))  int i32x4;

#define D_MODEL 1024
#define N_HEAD  16
#define HEAD_DIM 64
#define BATCH   4
#define SEQ     2048

__device__ __forceinline__ __bf16 to_b(float x)  { return (__bf16)x; }
__device__ __forceinline__ __bf16 to_b(__bf16 x) { return x; }

__device__ __forceinline__ v8f wmma_bf16(v16bf a, v16bf b, v8f c) {
  // D(16x16,f32) = A(16x32,bf16) * B(32x16,bf16) + C
  return __builtin_amdgcn_wmma_f32_16x16x32_bf16(false, a, false, b, (short)0, c,
                                                 false, false);
}

// Load one 16x32 bf16 operand fragment (A, or B mirrored with lane->N) from LDS.
// rowp points at k=0 of the row this lane owns (caller picks row via lane&15).
__device__ __forceinline__ v16bf load_frag(const __bf16* rowp) {
  const int koff = (threadIdx.x & 16) ? 8 : 0;
  const unsigned* p = reinterpret_cast<const unsigned*>(rowp);
  union { unsigned u[8]; v16bf v; } f;
#pragma unroll
  for (int j = 0; j < 4; ++j) {
    f.u[j]     = p[(koff >> 1) + j];        // K = koff .. koff+7
    f.u[4 + j] = p[((16 + koff) >> 1) + j]; // K = 16+koff .. 16+koff+7
  }
  return f.v;
}

// TDM: DMA a [rows x 64] bf16 tile (row stride D_MODEL elements) from global
// memory into LDS with hardware padding to a 72-element (144B) LDS row stride.
// D# built per CDNA5 ISA ch.8. Issue from one wave only; EXEC is ignored.
__device__ __forceinline__ void tdm_load_tile_bf16(unsigned lds_byte_addr,
                                                   const __bf16* gptr, int rows) {
  const unsigned long long ga = (unsigned long long)(uintptr_t)gptr;
  u32x4 g0;
  g0.x = 1u;                                   // count=1 (valid), no gather
  g0.y = lds_byte_addr;                        // lds_addr [63:32]
  g0.z = (unsigned)(ga & 0xffffffffu);         // global_addr[31:0]
  g0.w = ((unsigned)(ga >> 32) & 0x01ffffffu)  // global_addr[56:32]
         | (2u << 30);                         // type=2 ("image")
  i32x8 g1;
  g1[0] = (int)0x07110000u; // data_size=2B; pad_en; pad_interval=32DW; pad_amount=4DW
  g1[1] = (int)(HEAD_DIM << 16);   // tensor_dim0 = 64   (bits 79:48, low half)
  g1[2] = (int)(SEQ << 16);        // tensor_dim1 = 2048 (bits 111:80, low half)
  g1[3] = (int)(HEAD_DIM << 16);   // tile_dim0 = 64     (bits 127:112)
  g1[4] = rows;                    // tile_dim1          (bits 143:128)
  g1[5] = D_MODEL;                 // tensor_dim0_stride = 1024 elements
  g1[6] = 0;
  g1[7] = 0;
  const i32x4 gz = {0, 0, 0, 0};
#if __clang_major__ >= 23
  const i32x8 gz8 = {0, 0, 0, 0, 0, 0, 0, 0};
  __builtin_amdgcn_tensor_load_to_lds(g0, g1, gz, gz, gz8, 0);
#else
  __builtin_amdgcn_tensor_load_to_lds(g0, g1, gz, gz, 0);
#endif
}

// C = X[M,K] * W[K,N] + bias, K=N=1024. Block computes 128x128, 8 waves of 32x64.
template <typename TIn, typename TOut, bool NT_STORE>
__global__ __launch_bounds__(256) void gemm_bias_wmma(
    const TIn* __restrict__ X, const float* __restrict__ W,
    const float* __restrict__ bias, TOut* __restrict__ out, int M) {
  constexpr int N = D_MODEL, K = D_MODEL, BK = 32, LS = 36;
  __shared__ alignas(16) __bf16 Xs[128 * LS];   // [row][k]
  __shared__ alignas(16) __bf16 Wt[128 * LS];   // transposed: [n][k]
  const int t = threadIdx.x;
  const int lane = t & 31, wave = t >> 5;
  const int wm = wave & 3, wn = wave >> 2;
  const int l15 = lane & 15;
  const int row0 = blockIdx.y * 128;
  const int col0 = blockIdx.x * 128;
  const v8f vzero = {};
  v8f acc[2][4];
#pragma unroll
  for (int s = 0; s < 2; ++s)
#pragma unroll
    for (int nt = 0; nt < 4; ++nt) acc[s][nt] = vzero;

  for (int kc = 0; kc < K; kc += BK) {
#pragma unroll
    for (int i = 0; i < 16; ++i) {
      const int idx = t + i * 256;
      const int xr = idx >> 5, xc = idx & 31;
      Xs[xr * LS + xc] = to_b(X[(size_t)(row0 + xr) * K + kc + xc]);
      const int wr = idx >> 7, wc = idx & 127;
      Wt[wc * LS + wr] = to_b(W[(size_t)(kc + wr) * N + col0 + wc]);
    }
    if (kc + BK < K)
      __builtin_prefetch(&X[(size_t)(row0 + (t >> 1)) * K + kc + BK], 0, 1);
    __syncthreads();
    v16bf a0 = load_frag(&Xs[(wm * 32 + l15) * LS]);
    v16bf a1 = load_frag(&Xs[(wm * 32 + 16 + l15) * LS]);
#pragma unroll
    for (int nt = 0; nt < 4; ++nt) {
      v16bf bf = load_frag(&Wt[(wn * 64 + nt * 16 + l15) * LS]);
      acc[0][nt] = wmma_bf16(a0, bf, acc[0][nt]);
      acc[1][nt] = wmma_bf16(a1, bf, acc[1][nt]);
    }
    __syncthreads();
  }
  const int hi8 = (lane & 16) ? 8 : 0;
#pragma unroll
  for (int s = 0; s < 2; ++s)
#pragma unroll
    for (int nt = 0; nt < 4; ++nt)
#pragma unroll
      for (int r = 0; r < 8; ++r) {
        const int m = row0 + wm * 32 + s * 16 + r + hi8;
        const int n = col0 + wn * 64 + nt * 16 + l15;
        const TOut val = (TOut)(acc[s][nt][r] + bias[n]);
        if constexpr (NT_STORE)
          __builtin_nontemporal_store(val, &out[(size_t)m * N + n]);
        else
          out[(size_t)m * N + n] = val;
      }
}

// Flash attention: one block per (b, h, 128-query tile); 8 waves, 16 queries each.
// Q and K tiles staged by the Tensor Data Mover; V^T staged manually (transpose).
__global__ __launch_bounds__(256) void attn_fused_wmma(
    const __bf16* __restrict__ Q, const __bf16* __restrict__ Kp,
    const __bf16* __restrict__ V, const float* __restrict__ amask,
    __bf16* __restrict__ ctx) {
  constexpr int LS = 72;
  __shared__ alignas(16) __bf16 Qs[128 * LS];      // [q][d]     (TDM)
  __shared__ alignas(16) __bf16 Ks[64 * LS];       // [key][d]   (TDM)
  __shared__ alignas(16) __bf16 Vt[64 * LS];       // [d][key]   (manual transpose)
  __shared__ alignas(16) __bf16 Ps[8 * 16 * LS];   // per-wave P: [16 q][64 key]
  const int t = threadIdx.x;
  const int lane = t & 31, wave = t >> 5;
  const int l15 = lane & 15;
  const int hi8 = (lane & 16) ? 8 : 0;
  const int q0 = blockIdx.x * 128;
  const int h = blockIdx.y;
  const int b = blockIdx.z;
  const size_t headoff = (size_t)b * SEQ * D_MODEL + (size_t)h * HEAD_DIM;

  if (wave == 0)  // async DMA of the whole 128x64 Q tile
    tdm_load_tile_bf16((unsigned)(uintptr_t)&Qs[0],
                       Q + headoff + (size_t)q0 * D_MODEL, 128);

  const v8f vzero = {};
  float mrow[8], lrow[8];
  v8f o[4];
#pragma unroll
  for (int r = 0; r < 8; ++r) { mrow[r] = -3.0e38f; lrow[r] = 0.0f; }
#pragma unroll
  for (int nt = 0; nt < 4; ++nt) o[nt] = vzero;

  const int qmax = q0 + wave * 16 + 15;
  const int ntiles = (q0 >> 6) + 2;                // causal: keys <= q0+127
  for (int tk = 0; tk < ntiles; ++tk) {
    const int kv0 = tk * 64;
    if (wave == 0)  // async DMA of the 64x64 K tile (overlaps V^T staging)
      tdm_load_tile_bf16((unsigned)(uintptr_t)&Ks[0],
                         Kp + headoff + (size_t)kv0 * D_MODEL, 64);
#pragma unroll
    for (int i = 0; i < 16; ++i) {                 // stage 64x64 V^T
      const int idx = t + i * 256;
      const int r = idx >> 6, c = idx & 63;
      Vt[c * LS + r] = V[headoff + (size_t)(kv0 + r) * D_MODEL + c];
    }
    if (wave == 0) __builtin_amdgcn_s_wait_tensorcnt(0);
    __syncthreads();
    if (kv0 <= qmax) {                             // skip fully-masked tiles
      v8f sc[4];
#pragma unroll
      for (int nt = 0; nt < 4; ++nt) sc[nt] = vzero;
#pragma unroll
      for (int kc = 0; kc < 2; ++kc) {             // S = Q * K^T  (Dh=64 = 2x32)
        v16bf aq = load_frag(&Qs[(wave * 16 + l15) * LS + kc * 32]);
#pragma unroll
        for (int nt = 0; nt < 4; ++nt) {
          v16bf bk = load_frag(&Ks[(nt * 16 + l15) * LS + kc * 32]);
          sc[nt] = wmma_bf16(aq, bk, sc[nt]);
        }
      }
#pragma unroll
      for (int nt = 0; nt < 4; ++nt) {             // scale + pad mask + causal
        const int kg = kv0 + nt * 16 + l15;
        const float am = amask[(size_t)b * SEQ + kg];
#pragma unroll
        for (int r = 0; r < 8; ++r) {
          const int qg = q0 + wave * 16 + r + hi8;
          const float val = sc[nt][r] * 0.125f + am;
          sc[nt][r] = (kg > qg) ? -3.0e38f : val;
        }
      }
      __bf16* pw = &Ps[wave * 16 * LS];            // online softmax (16-lane halves)
#pragma unroll
      for (int r = 0; r < 8; ++r) {
        float mx = sc[0][r];
#pragma unroll
        for (int nt = 1; nt < 4; ++nt) mx = fmaxf(mx, sc[nt][r]);
#pragma unroll
        for (int msk = 8; msk >= 1; msk >>= 1) mx = fmaxf(mx, __shfl_xor(mx, msk, 32));
        const float mnew = fmaxf(mrow[r], mx);
        const float sf = __expf(mrow[r] - mnew);
        mrow[r] = mnew;
        float rs = 0.0f;
#pragma unroll
        for (int nt = 0; nt < 4; ++nt) {
          const float p = __expf(sc[nt][r] - mnew);
          rs += p;
          o[nt][r] = o[nt][r] * sf;
          pw[(r + hi8) * LS + nt * 16 + l15] = (__bf16)p;  // C-layout -> LDS
        }
#pragma unroll
        for (int msk = 8; msk >= 1; msk >>= 1) rs += __shfl_xor(rs, msk, 32);
        lrow[r] = lrow[r] * sf + rs;
      }
      asm volatile("s_wait_dscnt 0" ::: "memory"); // intra-wave LDS RAW on Ps
#pragma unroll
      for (int kc = 0; kc < 2; ++kc) {             // O += P * V
        v16bf ap = load_frag(&pw[l15 * LS + kc * 32]);
#pragma unroll
        for (int nt = 0; nt < 4; ++nt) {
          v16bf bv = load_frag(&Vt[(nt * 16 + l15) * LS + kc * 32]);
          o[nt] = wmma_bf16(ap, bv, o[nt]);
        }
      }
    }
    __syncthreads();  // also protects Ks against next iteration's TDM issue
  }
#pragma unroll
  for (int r = 0; r < 8; ++r) {                    // epilogue: O / l
    const float inv = 1.0f / lrow[r];
    const int qg = q0 + wave * 16 + r + hi8;
#pragma unroll
    for (int nt = 0; nt < 4; ++nt)
      ctx[headoff + (size_t)qg * D_MODEL + nt * 16 + l15] = (__bf16)(o[nt][r] * inv);
  }
}

extern "C" void kernel_launch(void* const* d_in, const int* in_sizes, int n_in,
                              void* d_out, int out_size, void* d_ws, size_t ws_size,
                              hipStream_t stream) {
  (void)in_sizes; (void)n_in; (void)out_size; (void)ws_size;
  const float* q  = (const float*)d_in[0];
  const float* k  = (const float*)d_in[1];
  const float* v  = (const float*)d_in[2];
  const float* am = (const float*)d_in[3];
  const float* Wq = (const float*)d_in[4];
  const float* bq = (const float*)d_in[5];
  const float* Wk = (const float*)d_in[6];
  const float* bk = (const float*)d_in[7];
  const float* Wv = (const float*)d_in[8];
  const float* bv = (const float*)d_in[9];
  const float* Wp = (const float*)d_in[10];
  const float* bp = (const float*)d_in[11];
  float* out = (float*)d_out;

  const size_t nElem = (size_t)BATCH * SEQ * D_MODEL;   // 8M elems; ws use: 64 MiB
  __bf16* Qp  = reinterpret_cast<__bf16*>(d_ws);
  __bf16* Kpj = Qp + nElem;
  __bf16* Vp  = Kpj + nElem;
  __bf16* Cx  = Vp + nElem;

  const int M = BATCH * SEQ;                            // 8192
  dim3 ggrid(D_MODEL / 128, M / 128);                   // (8, 64)
  dim3 blk(256);
  gemm_bias_wmma<float, __bf16, false><<<ggrid, blk, 0, stream>>>(q, Wq, bq, Qp, M);
  gemm_bias_wmma<float, __bf16, false><<<ggrid, blk, 0, stream>>>(k, Wk, bk, Kpj, M);
  gemm_bias_wmma<float, __bf16, false><<<ggrid, blk, 0, stream>>>(v, Wv, bv, Vp, M);

  dim3 agrid(SEQ / 128, N_HEAD, BATCH);                 // (16, 16, 4)
  attn_fused_wmma<<<agrid, blk, 0, stream>>>(Qp, Kpj, Vp, am, Cx);

  gemm_bias_wmma<__bf16, float, true><<<ggrid, blk, 0, stream>>>(Cx, Wp, bp, out, M);
}
